// CNN_medium_283467841699
// MI455X (gfx1250) — 8-bit matrix operand order repaired and checked by reading, compiled, not run on hardware
//
#include <hip/hip_runtime.h>
#include <stdint.h>

typedef __attribute__((ext_vector_type(8))) int v8i;

#define BATCH 128

static const size_t OFF_ACT_A = 0;
static const size_t OFF_ACT_B = (size_t)32 << 20;
static const size_t OFF_RAW   = (size_t)64 << 20;
static const size_t OFF_WB    = (size_t)128 << 20;

__device__ __forceinline__ int8_t sgn8(float z) {
    return (z > 0.f) ? (int8_t)1 : ((z < 0.f) ? (int8_t)-1 : (int8_t)0);
}

__global__ void k_binarize_w(const float* __restrict__ w, int8_t* __restrict__ out,
                             int O, int I, int KH, int KW, int Opad)
{
    int idx = blockIdx.x * blockDim.x + threadIdx.x;
    int total = Opad * KH * KW * I;
    if (idx >= total) return;
    int c = idx % I;
    int t = idx / I;
    int kw = t % KW; t /= KW;
    int kh = t % KH; t /= KH;
    int o  = t;
    int8_t v = 0;
    if (o < O) {
        float f = w[((o * I + c) * KH + kh) * KW + kw];
        v = (f > 0.f) ? 1 : ((f < 0.f) ? -1 : 0);
    }
    out[idx] = v;
}

__global__ void k_conv1(const float* __restrict__ x, const int8_t* __restrict__ wb,
                        const float* __restrict__ s, const float* __restrict__ t,
                        int8_t* __restrict__ act)
{
    int idx = blockIdx.x * blockDim.x + threadIdx.x;
    int o  = idx & 127;
    int px = idx >> 7;
    int ox = px & 31;
    int oy = (px >> 5) & 31;
    int n  = px >> 10;
    float acc = 0.f;
    #pragma unroll
    for (int kh = 0; kh < 3; ++kh) {
        int iy = oy + kh - 1;
        if ((unsigned)iy >= 32u) continue;
        #pragma unroll
        for (int kw = 0; kw < 3; ++kw) {
            int ix = ox + kw - 1;
            if ((unsigned)ix >= 32u) continue;
            #pragma unroll
            for (int c = 0; c < 3; ++c) {
                float xv = x[((n * 3 + c) * 32 + iy) * 32 + ix];
                float wv = (float)wb[o * 27 + (kh * 3 + kw) * 3 + c];
                acc += xv * wv;
            }
        }
    }
    float z = s[o] * acc + t[o];
    act[(size_t)px * 128 + o] = sgn8(z);
}

template<int CIN, int COUT, int S, int FUSE>
__global__ void __launch_bounds__(256)
k_bconv(const int8_t* __restrict__ actin, const int8_t* __restrict__ wgt,
        const float* __restrict__ s, const float* __restrict__ t,
        int8_t* __restrict__ actout, int* __restrict__ raw)
{
    constexpr int SS = S * S;
    __shared__ int4 Bsm[(32 * CIN) / 16];

    const int tid   = threadIdx.x;
    const int lane  = tid & 31;
    const int wv    = tid >> 5;
    const int mbase = (blockIdx.y * 8 + wv) * 16;

    const int spix = tid >> 3;
    const int q    = tid & 7;
    const int sp   = blockIdx.x * 32 + spix;
    const int sn   = sp / SS;
    const int srem = sp % SS;
    const int soy  = srem / S;
    const int sox  = srem % S;
    int4* myLds = &Bsm[spix * (CIN / 16) + q * (CIN / 128)];
    const unsigned myLdsAddr = (unsigned)(unsigned long long)myLds;

    const int mrow  = lane & 15;
    const int khalf = lane >> 4;
    const int nlane = lane & 15;
    const int8_t* arow = wgt + (mbase + mrow) * (9 * CIN) + khalf * 16;

    v8i acc[2];
    acc[0] = (v8i){0, 0, 0, 0, 0, 0, 0, 0};
    acc[1] = (v8i){0, 0, 0, 0, 0, 0, 0, 0};

    #pragma unroll 1
    for (int kh = 0; kh < 3; ++kh) {
        const int iy = soy + kh - 1;
        #pragma unroll 1
        for (int kw = 0; kw < 3; ++kw) {
            const int ix = sox + kw - 1;
            const bool valid = ((unsigned)iy < (unsigned)S) && ((unsigned)ix < (unsigned)S);
            const int8_t* src = actin + ((long)((sn * S + iy) * S + ix)) * CIN + q * (CIN / 8);

            __syncthreads();
            if (valid) {
                #pragma unroll
                for (int i = 0; i < CIN / 128; ++i) {
                    unsigned long long ga = (unsigned long long)(src + i * 16);
                    unsigned la = myLdsAddr + i * 16;
                    asm volatile("global_load_async_to_lds_b128 %0, %1, off"
                                 :: "v"(la), "v"(ga) : "memory");
                }
            } else {
                #pragma unroll
                for (int i = 0; i < CIN / 128; ++i)
                    myLds[i] = make_int4(0, 0, 0, 0);
            }
            asm volatile("s_wait_asynccnt 0x0" ::: "memory");
            __syncthreads();

            const int kpos = kh * 3 + kw;
            const int8_t* ap = arow + kpos * CIN;
            __builtin_prefetch(arow + (kpos + 1) * CIN, 0, 0);

            #pragma unroll
            for (int c0 = 0; c0 < CIN; c0 += 64) {
                v8i a;
                {
                    int4 w0 = *(const int4*)(ap + c0 + 0);
                    int4 w1 = *(const int4*)(ap + c0 + 32);
                    a[0] = w0.x; a[1] = w0.y; a[2] = w0.z; a[3] = w0.w;
                    a[4] = w1.x; a[5] = w1.y; a[6] = w1.z; a[7] = w1.w;
                }
                #pragma unroll
                for (int nt = 0; nt < 2; ++nt) {
                    const int bi = (nt * 16 + nlane) * (CIN / 16) + (c0 >> 4) + khalf;
                    int4 b0 = Bsm[bi];
                    int4 b1 = Bsm[bi + 2];
                    v8i b;
                    b[0] = b0.x; b[1] = b0.y; b[2] = b0.z; b[3] = b0.w;
                    b[4] = b1.x; b[5] = b1.y; b[6] = b1.z; b[7] = b1.w;
                    acc[nt] = __builtin_amdgcn_wmma_i32_16x16x64_iu8(
                        true, a, true, b, acc[nt], false, false);
                }
            }
        }
    }

    #pragma unroll
    for (int nt = 0; nt < 2; ++nt) {
        const long p = (long)blockIdx.x * 32 + nt * 16 + nlane;
        const int  moff = mbase + khalf * 8;
        if (FUSE) {
            unsigned long long pack = 0;
            #pragma unroll
            for (int r = 0; r < 8; ++r) {
                int m = moff + r;
                float z = s[m] * (float)acc[nt][r] + t[m];
                pack |= (unsigned long long)(unsigned char)sgn8(z) << (8 * r);
            }
            *(unsigned long long*)(actout + p * COUT + moff) = pack;
        } else {
            int* dst = raw + p * COUT + moff;
            *(int4*)(dst)     = make_int4(acc[nt][0], acc[nt][1], acc[nt][2], acc[nt][3]);
            *(int4*)(dst + 4) = make_int4(acc[nt][4], acc[nt][5], acc[nt][6], acc[nt][7]);
        }
    }
}

template<int C, int SIN>
__global__ void k_pool_bn_sign(const int* __restrict__ raw,
                               const float* __restrict__ s, const float* __restrict__ t,
                               int8_t* __restrict__ out)
{
    constexpr int SO = SIN / 2;
    long idx   = (long)blockIdx.x * blockDim.x + threadIdx.x;
    long total = (long)BATCH * SO * SO * C;
    if (idx >= total) return;
    int  c  = (int)(idx % C);
    long pp = idx / C;
    int  x  = (int)(pp % SO);
    int  y  = (int)((pp / SO) % SO);
    int  n  = (int)(pp / (SO * SO));
    long b  = (((long)n * SIN + 2 * y) * SIN + 2 * x) * C + c;
    int v0 = raw[b];
    int v1 = raw[b + C];
    int v2 = raw[b + (long)SIN * C];
    int v3 = raw[b + (long)SIN * C + C];
    int m01 = v0 > v1 ? v0 : v1;
    int m23 = v2 > v3 ? v2 : v3;
    int mx  = m01 > m23 ? m01 : m23;
    float z = s[c] * (float)mx + t[c];
    out[idx] = sgn8(z);
}

__global__ void __launch_bounds__(256)
k_conv7_wmma(const int8_t* __restrict__ act,
             const int8_t* __restrict__ wgt,
             int* __restrict__ logits)
{
    const int tid   = threadIdx.x;
    const int lane  = tid & 31;
    const int wv    = tid >> 5;
    const int mrow  = lane & 15;
    const int khalf = lane >> 4;
    const int nlane = lane & 15;
    const int K = 4 * 4 * 512;

    v8i acc = {0, 0, 0, 0, 0, 0, 0, 0};
    const int8_t* ap0 = wgt + (long)mrow * K + khalf * 16;
    const int8_t* bp0 = act + (long)(wv * 16 + nlane) * K + khalf * 16;

    #pragma unroll 4
    for (int c0 = 0; c0 < K; c0 += 64) {
        v8i a, b;
        {
            int4 w0 = *(const int4*)(ap0 + c0 + 0);
            int4 w1 = *(const int4*)(ap0 + c0 + 32);
            a[0] = w0.x; a[1] = w0.y; a[2] = w0.z; a[3] = w0.w;
            a[4] = w1.x; a[5] = w1.y; a[6] = w1.z; a[7] = w1.w;
        }
        {
            int4 b0 = *(const int4*)(bp0 + c0 + 0);
            int4 b1 = *(const int4*)(bp0 + c0 + 32);
            b[0] = b0.x; b[1] = b0.y; b[2] = b0.z; b[3] = b0.w;
            b[4] = b1.x; b[5] = b1.y; b[6] = b1.z; b[7] = b1.w;
        }
        acc = __builtin_amdgcn_wmma_i32_16x16x64_iu8(true, a, true, b, acc,
                                                     false, false);
    }
    int* dst = logits + (wv * 16 + nlane) * 16 + khalf * 8;
    *(int4*)(dst)     = make_int4(acc[0], acc[1], acc[2], acc[3]);
    *(int4*)(dst + 4) = make_int4(acc[4], acc[5], acc[6], acc[7]);
}

__global__ void k_bn_logsoftmax(const int* __restrict__ logits,
                                const float* __restrict__ s, const float* __restrict__ t,
                                float* __restrict__ out)
{
    int n = blockIdx.x * blockDim.x + threadIdx.x;
    if (n >= BATCH) return;
    float z[10];
    float mx = -1e30f;
    #pragma unroll
    for (int j = 0; j < 10; ++j) {
        z[j] = s[j] * (float)logits[n * 16 + j] + t[j];
        mx = fmaxf(mx, z[j]);
    }
    float sum = 0.f;
    #pragma unroll
    for (int j = 0; j < 10; ++j) sum += __expf(z[j] - mx);
    float lse = mx + __logf(sum);
    #pragma unroll
    for (int j = 0; j < 10; ++j) out[n * 10 + j] = z[j] - lse;
}

extern "C" void kernel_launch(void* const* d_in, const int* in_sizes, int n_in,
                              void* d_out, int out_size, void* d_ws, size_t ws_size,
                              hipStream_t stream)
{
    const float* x = (const float*)d_in[0];
    const float* w[7];
    const float* bns[7];
    const float* bnt[7];
    for (int i = 0; i < 7; ++i) w[i] = (const float*)d_in[1 + i];
    for (int i = 0; i < 7; ++i) {
        bns[i] = (const float*)d_in[8 + 2 * i];
        bnt[i] = (const float*)d_in[9 + 2 * i];
    }

    int8_t* ws8  = (int8_t*)d_ws;
    int8_t* actA = ws8 + OFF_ACT_A;
    int8_t* actB = ws8 + OFF_ACT_B;
    int*    raw  = (int*)(ws8 + OFF_RAW);
    int8_t* wb   = ws8 + OFF_WB;

    const int O[7]    = {128, 128, 256, 256, 512, 512, 10};
    const int I[7]    = {3,   128, 128, 256, 256, 512, 512};
    const int KH[7]   = {3, 3, 3, 3, 3, 3, 4};
    const int Opad[7] = {128, 128, 256, 256, 512, 512, 16};

    size_t wOff[7];
    size_t wAcc = 0;
    for (int i = 0; i < 7; ++i) {
        wOff[i] = wAcc;
        wAcc += (size_t)Opad[i] * KH[i] * KH[i] * I[i];
    }
    int* logits = (int*)(ws8 + OFF_WB + ((wAcc + 255) & ~(size_t)255));

    for (int i = 0; i < 7; ++i) {
        int total = Opad[i] * KH[i] * KH[i] * I[i];
        k_binarize_w<<<(total + 255) / 256, 256, 0, stream>>>(
            w[i], wb + wOff[i], O[i], I[i], KH[i], KH[i], Opad[i]);
    }

    k_conv1<<<(BATCH * 32 * 32 * 128) / 256, 256, 0, stream>>>(
        x, wb + wOff[0], bns[0], bnt[0], actA);

    k_bconv<128, 128, 32, 0><<<dim3(BATCH * 32 * 32 / 32, 1), 256, 0, stream>>>(
        actA, wb + wOff[1], nullptr, nullptr, nullptr, raw);
    k_pool_bn_sign<128, 32><<<(BATCH * 16 * 16 * 128 + 255) / 256, 256, 0, stream>>>(
        raw, bns[1], bnt[1], actB);

    k_bconv<128, 256, 16, 1><<<dim3(BATCH * 16 * 16 / 32, 2), 256, 0, stream>>>(
        actB, wb + wOff[2], bns[2], bnt[2], actA, nullptr);

    k_bconv<256, 256, 16, 0><<<dim3(BATCH * 16 * 16 / 32, 2), 256, 0, stream>>>(
        actA, wb + wOff[3], nullptr, nullptr, nullptr, raw);
    k_pool_bn_sign<256, 16><<<(BATCH * 8 * 8 * 256 + 255) / 256, 256, 0, stream>>>(
        raw, bns[3], bnt[3], actB);

    k_bconv<256, 512, 8, 1><<<dim3(BATCH * 8 * 8 / 32, 4), 256, 0, stream>>>(
        actB, wb + wOff[4], bns[4], bnt[4], actA, nullptr);

    k_bconv<512, 512, 8, 0><<<dim3(BATCH * 8 * 8 / 32, 4), 256, 0, stream>>>(
        actA, wb + wOff[5], nullptr, nullptr, nullptr, raw);
    k_pool_bn_sign<512, 8><<<(BATCH * 4 * 4 * 512 + 255) / 256, 256, 0, stream>>>(
        raw, bns[5], bnt[5], actB);

    k_conv7_wmma<<<1, 256, 0, stream>>>(actB, wb + wOff[6], logits);

    k_bn_logsoftmax<<<1, 128, 0, stream>>>(logits, bns[6], bnt[6], (float*)d_out);
}
